// QKVAttention_1597727834254
// MI455X (gfx1250) — matrix operand order repaired and checked by reading, not compiled, not run on hardware
//
#include <hip/hip_runtime.h>

typedef __attribute__((ext_vector_type(16))) _Float16 v16h;
typedef __attribute__((ext_vector_type(8)))  _Float16 v8h;
typedef __attribute__((ext_vector_type(4)))  _Float16 v4h;
typedef __attribute__((ext_vector_type(8)))  float    v8f;
typedef __attribute__((ext_vector_type(4)))  float    v4f;

union AF { v16h v; v8h h[2]; };
union SF { v8f v; float f[8]; };
union BF { v16h v; unsigned u[8]; };
union H2 { _Float16 h[2]; unsigned u; };

#define B_   4
#define L_   2048
#define C_   512
#define H_   8
#define CH_  64
#define BH_  (B_*H_)
#define M_   (B_*L_)     /* 8192 */
#define N3_  (3*C_)      /* 1536 */
#define NT_  (L_/64)     /* 32 key tiles */
#define QK_SCALE 0.35355339059327373f  /* 1/sqrt(sqrt(64)) */

// ---------------------------------------------------------------------------
// Kernel 1: qkv = x @ W + b -> head-split f16 Q (scaled) [bh][t][ch],
//           K (scaled) [bh][t][ch], V [bh][ch][t] (pre-transposed for attn).
// Block: 128 threads (4 waves), 64x64 output tile, 2x2 WMMA per wave.
// ---------------------------------------------------------------------------
__global__ __launch_bounds__(128)
void qkv_proj_kernel(const float* __restrict__ x, const float* __restrict__ W,
                     const float* __restrict__ bias,
                     _Float16* __restrict__ Qh, _Float16* __restrict__ Kh,
                     _Float16* __restrict__ Vh)
{
    __shared__ __align__(64) _Float16 Alds[64*32];   // [m][k]
    __shared__ __align__(64) _Float16 Blds[64*32];   // transposed: [n][k]

    const int tid   = threadIdx.x;
    const int lane  = tid & 31;
    const int wv    = tid >> 5;
    const int lh    = lane >> 4;
    const int ll    = lane & 15;
    const int mBase = blockIdx.y * 64;
    const int nBase = blockIdx.x * 64;
    const int mo    = (wv >> 1) * 32;
    const int no    = (wv & 1) * 32;

    SF acc[2][2];
    #pragma unroll
    for (int i = 0; i < 2; ++i)
        #pragma unroll
        for (int j = 0; j < 2; ++j)
            #pragma unroll
            for (int e = 0; e < 8; ++e) acc[i][j].f[e] = 0.0f;

    for (int k0 = 0; k0 < C_; k0 += 32) {
        // Batch all 8 global loads first: deep outstanding-load window.
        v4f ar[4], br[4];
        #pragma unroll
        for (int it = 0; it < 4; ++it) {
            int idx = tid + it * 128;                 // A: 64 rows x 8 chunks
            ar[it] = *(const v4f*)(x + (size_t)(mBase + (idx >> 3)) * C_
                                     + k0 + ((idx & 7) << 2));
        }
        #pragma unroll
        for (int it = 0; it < 4; ++it) {
            int idx = tid + it * 128;                 // B: 32 rows x 16 chunks
            br[it] = *(const v4f*)(W + (size_t)(k0 + (idx >> 4)) * N3_
                                     + nBase + ((idx & 15) << 2));
        }
        __syncthreads();   // previous iteration's fragment reads finished
        #pragma unroll
        for (int it = 0; it < 4; ++it) {
            int idx = tid + it * 128;
            int row = idx >> 3, c4 = (idx & 7) << 2;
            v4h ah;
            ah.x = (_Float16)ar[it].x; ah.y = (_Float16)ar[it].y;
            ah.z = (_Float16)ar[it].z; ah.w = (_Float16)ar[it].w;
            *(v4h*)(&Alds[row * 32 + c4]) = ah;
        }
        #pragma unroll
        for (int it = 0; it < 4; ++it) {
            int idx = tid + it * 128;
            int kr = idx >> 4, c4 = (idx & 15) << 2;
            Blds[(c4 + 0) * 32 + kr] = (_Float16)br[it].x;
            Blds[(c4 + 1) * 32 + kr] = (_Float16)br[it].y;
            Blds[(c4 + 2) * 32 + kr] = (_Float16)br[it].z;
            Blds[(c4 + 3) * 32 + kr] = (_Float16)br[it].w;
        }
        __syncthreads();

        AF a[2], b[2];
        #pragma unroll
        for (int mt = 0; mt < 2; ++mt) {
            int row = mo + mt * 16 + ll;
            int kb  = lh * 16;                         // contiguous k = 16*lh + s
            a[mt].v = *(const v16h*)(&Alds[row * 32 + kb]);
        }
        #pragma unroll
        for (int nt = 0; nt < 2; ++nt) {
            int n = no + nt * 16 + ll;                 // contiguous k = 16*lh + s, as A
            b[nt].v = *(const v16h*)(&Blds[n * 32 + lh * 16]);
        }
        #pragma unroll
        for (int mt = 0; mt < 2; ++mt)
            #pragma unroll
            for (int nt = 0; nt < 2; ++nt)
                acc[mt][nt].v = __builtin_amdgcn_wmma_f32_16x16x32_f16(
                    false, a[mt].v, false, b[nt].v, (short)0, acc[mt][nt].v,
                    false, false);
    }

    // Epilogue: bias + scale, scatter to Q/K ([bh][t][ch]) and V ([bh][ch][t])
    #pragma unroll
    for (int nt = 0; nt < 2; ++nt) {
        int   c    = nBase + no + nt * 16 + ll;
        float bv   = bias[c];
        int   sec  = c / C_;                 // 0=q 1=k 2=v
        int   cc   = c - sec * C_;
        int   head = cc >> 6;
        int   chan = cc & 63;
        float sc = (sec == 2) ? 1.0f : QK_SCALE;
        #pragma unroll
        for (int mt = 0; mt < 2; ++mt) {
            #pragma unroll
            for (int j = 0; j < 8; ++j) {
                int m    = mBase + mo + mt * 16 + j + 8 * lh;  // C-layout row
                int bidx = m >> 11;
                int t    = m & (L_ - 1);
                int bh   = bidx * H_ + head;
                float v  = (acc[mt][nt].f[j] + bv) * sc;
                if (sec == 0)
                    Qh[((size_t)bh * L_ + t) * CH_ + chan] = (_Float16)v;
                else if (sec == 1)
                    Kh[((size_t)bh * L_ + t) * CH_ + chan] = (_Float16)v;
                else  // V stored channel-major: [bh][ch][t]
                    Vh[((size_t)bh * CH_ + chan) * L_ + t] = (_Float16)v;
            }
        }
    }
}

// ---------------------------------------------------------------------------
// Kernel 2: flash attention in transposed orientation:
//   S^T = K  . Q^T   (A = K tile  [key][ch],  B = Q^T  from registers)
//   O^T = V^T. P^T   (A = V^T tile [ch][key], B = P^T  built via shfl_xor 16)
// Each lane owns exactly one query column -> softmax is in-lane + 1 shuffle.
// K/V tiles staged with gfx1250 async DMA (ASYNCcnt), double buffered.
// Block: 128 threads (4 waves); wave w owns query rows qBase+16w..+15.
// ---------------------------------------------------------------------------
__global__ __launch_bounds__(128)
void flash_attn_kernel(const _Float16* __restrict__ Qh,
                       const _Float16* __restrict__ Kh,
                       const _Float16* __restrict__ Vh,
                       float* __restrict__ out)
{
    __shared__ __align__(64) _Float16 Klds[2][64*64];   // [key][ch]
    __shared__ __align__(64) _Float16 Vlds[2][64*64];   // [ch][key]

    const int tid   = threadIdx.x;
    const int lane  = tid & 31;
    const int wv    = tid >> 5;
    const int lh    = lane >> 4;
    const int ll    = lane & 15;
    const int bh    = blockIdx.y;
    const int qBase = blockIdx.x * 64;
    const int bidx  = bh >> 3;
    const int head  = bh & 7;

    const _Float16* Qb = Qh + (size_t)bh * L_ * CH_;   // [t][ch]
    const _Float16* Kb = Kh + (size_t)bh * L_ * CH_;   // [t][ch]
    const _Float16* Vb = Vh + (size_t)bh * CH_ * L_;   // [ch][t]

    // Per-thread async-copy slot (4 x 16-byte chunks per matrix per tile)
    const int crow = tid >> 3;
    const int cc8  = (tid & 7) << 3;

    // Issue one key-tile's async DMA: 8 wave-instructions -> ASYNCcnt += 8
    auto issue_tile = [&](int kt, int buf) {
        const _Float16* Kg = Kb + (size_t)kt * 64 * CH_;   // 64 key rows
        const _Float16* Vg = Vb + (size_t)kt * 64;         // 64 t-columns
        #pragma unroll
        for (int it = 0; it < 4; ++it) {
            int r = crow + it * 16;                        // 0..63
            unsigned long long gk = (unsigned long long)(Kg + (size_t)r * CH_ + cc8);
            unsigned long long gv = (unsigned long long)(Vg + (size_t)r * L_  + cc8);
            unsigned lk = (unsigned)(unsigned long long)&Klds[buf][r * 64 + cc8];
            unsigned lv = (unsigned)(unsigned long long)&Vlds[buf][r * 64 + cc8];
            asm volatile("global_load_async_to_lds_b128 %0, %1, off"
                         :: "v"(lk), "v"(gk) : "memory");
            asm volatile("global_load_async_to_lds_b128 %0, %1, off"
                         :: "v"(lv), "v"(gv) : "memory");
        }
    };

    // Q^T B-fragments, resident in registers: b[h] = Q[q=ll][ch = kc*32+16*lh+h]
    v16h bq[2];
    {
        const _Float16* qrow = Qb + (size_t)(qBase + wv * 16 + ll) * CH_;
        bq[0] = *(const v16h*)(qrow + lh * 16);
        bq[1] = *(const v16h*)(qrow + 32 + lh * 16);
    }

    float m_i = -3.0e38f, l_i = 0.0f;
    SF o[4];
    #pragma unroll
    for (int mt = 0; mt < 4; ++mt)
        #pragma unroll
        for (int e = 0; e < 8; ++e) o[mt].f[e] = 0.0f;

    issue_tile(0, 0);                                   // prologue DMA

    for (int kt = 0; kt < NT_; ++kt) {
        const int buf = kt & 1;
        __syncthreads();    // all waves done reading buf^1 (tile kt-1)
        if (kt + 1 < NT_) {
            issue_tile(kt + 1, buf ^ 1);                // overlap DMA w/ compute
            asm volatile("s_wait_asynccnt 0x8" ::: "memory");  // tile kt landed
        } else {
            asm volatile("s_wait_asynccnt 0x0" ::: "memory");
        }
        __syncthreads();    // tile kt visible to all waves

        // S^T = K . Q^T : 64 keys (4 M-tiles) x 16 queries, 8 WMMAs
        const _Float16* Kbase = &Klds[buf][0];
        SF st[4];
        #pragma unroll
        for (int mt = 0; mt < 4; ++mt)
            #pragma unroll
            for (int e = 0; e < 8; ++e) st[mt].f[e] = 0.0f;
        #pragma unroll
        for (int mt = 0; mt < 4; ++mt) {
            #pragma unroll
            for (int kc = 0; kc < 2; ++kc) {
                AF ak;
                int off = (mt * 16 + ll) * 64 + kc * 32 + lh * 16;  // contiguous k = 16*lh + s
                ak.v = *(const v16h*)(Kbase + off);
                st[mt].v = __builtin_amdgcn_wmma_f32_16x16x32_f16(
                    false, ak.v, false, bq[kc], (short)0, st[mt].v, false, false);
            }
        }

        // Online softmax: this lane owns query (qBase+16w+ll); its 32 values
        // cover keys {j+8*lh} of each tile, partner lane (xor 16) has the rest.
        float r = st[0].f[0];
        #pragma unroll
        for (int mt = 0; mt < 4; ++mt)
            #pragma unroll
            for (int j = 0; j < 8; ++j) r = fmaxf(r, st[mt].f[j]);
        r = fmaxf(r, __shfl_xor(r, 16, 32));
        float newm = fmaxf(m_i, r);
        float fac  = __expf(m_i - newm);
        float rs   = 0.0f;
        #pragma unroll
        for (int mt = 0; mt < 4; ++mt)
            #pragma unroll
            for (int j = 0; j < 8; ++j) {
                float p = __expf(st[mt].f[j] - newm);
                st[mt].f[j] = p;
                rs += p;
            }
        rs += __shfl_xor(rs, 16, 32);
        l_i = l_i * fac + rs;
        m_i = newm;
        #pragma unroll
        for (int mt = 0; mt < 4; ++mt)
            #pragma unroll
            for (int j = 0; j < 8; ++j) o[mt].f[j] *= fac;

        // Build P^T B-fragments in registers: pack half2 words, swap halves
        // across the lh boundary with one shfl_xor(16) per word.
        unsigned own[4][4], oth[4][4];
        #pragma unroll
        for (int mt = 0; mt < 4; ++mt)
            #pragma unroll
            for (int d = 0; d < 4; ++d) {
                H2 t;
                t.h[0] = (_Float16)st[mt].f[2 * d];
                t.h[1] = (_Float16)st[mt].f[2 * d + 1];
                own[mt][d] = t.u;
            }
        #pragma unroll
        for (int mt = 0; mt < 4; ++mt)
            #pragma unroll
            for (int d = 0; d < 4; ++d)
                oth[mt][d] = (unsigned)__shfl_xor((int)own[mt][d], 16, 32);

        BF pb[2];
        #pragma unroll
        for (int kc = 0; kc < 2; ++kc) {
            // This lane's B-frag covers keys kc*32+16*lh .. +15 (tile T=kc*2+lh):
            //   keys 0-7 of T live in lh=0 lanes, keys 8-15 in lh=1 lanes.
            #pragma unroll
            for (int d = 0; d < 4; ++d) {
                pb[kc].u[d]     = lh ? oth[kc * 2 + 1][d] : own[kc * 2][d];
                pb[kc].u[4 + d] = lh ? own[kc * 2 + 1][d] : oth[kc * 2][d];
            }
        }

        // O^T += V^T . P^T : 64 channels (4 M-tiles) x 16 queries, 8 WMMAs
        const _Float16* Vbase = &Vlds[buf][0];
        #pragma unroll
        for (int mt = 0; mt < 4; ++mt) {
            #pragma unroll
            for (int kc = 0; kc < 2; ++kc) {
                AF av;
                int off = (mt * 16 + ll) * 64 + kc * 32 + lh * 16;  // contiguous k = 16*lh + s
                av.v = *(const v16h*)(Vbase + off);
                o[mt].v = __builtin_amdgcn_wmma_f32_16x16x32_f16(
                    false, av.v, false, pb[kc].v, (short)0, o[mt].v, false, false);
            }
        }
    }

    // Normalize + store: lane owns one output row t, contiguous 8-ch runs.
    float inv = 1.0f / l_i;
    int t = qBase + wv * 16 + ll;
    float* orow = out + ((size_t)bidx * L_ + t) * C_ + head * CH_;
    #pragma unroll
    for (int mt = 0; mt < 4; ++mt) {
        v4f o0, o1;
        o0.x = o[mt].f[0] * inv; o0.y = o[mt].f[1] * inv;
        o0.z = o[mt].f[2] * inv; o0.w = o[mt].f[3] * inv;
        o1.x = o[mt].f[4] * inv; o1.y = o[mt].f[5] * inv;
        o1.z = o[mt].f[6] * inv; o1.w = o[mt].f[7] * inv;
        *(v4f*)(orow + mt * 16 + lh * 8)     = o0;   // ch = mt*16 + 8*lh + j
        *(v4f*)(orow + mt * 16 + lh * 8 + 4) = o1;
    }
}

// ---------------------------------------------------------------------------
extern "C" void kernel_launch(void* const* d_in, const int* in_sizes, int n_in,
                              void* d_out, int out_size, void* d_ws, size_t ws_size,
                              hipStream_t stream)
{
    const float* x  = (const float*)d_in[0];
    const float* Wq = (const float*)d_in[1];
    const float* bq = (const float*)d_in[2];
    float* out = (float*)d_out;

    _Float16* Qh = (_Float16*)d_ws;                       // [BH][L][64]  (scaled)
    _Float16* Kh = Qh + (size_t)BH_ * L_ * CH_;           // [BH][L][64]  (scaled)
    _Float16* Vh = Kh + (size_t)BH_ * L_ * CH_;           // [BH][64][L]  (transposed)

    dim3 grid1(N3_ / 64, M_ / 64);                        // 24 x 128 blocks
    qkv_proj_kernel<<<grid1, 128, 0, stream>>>(x, Wq, bq, Qh, Kh, Vh);

    dim3 grid2(L_ / 64, BH_);                             // 32 x 32 blocks
    flash_attn_kernel<<<grid2, 128, 0, stream>>>(Qh, Kh, Vh, out);
}
